// GPTBlock_65360812310820
// MI455X (gfx1250) — matrix operand order repaired and checked by reading, not compiled, not run on hardware
//
#include <hip/hip_runtime.h>

// ---------------------------------------------------------------------------
// GPT block forward for MI455X (gfx1250, wave32, WMMA + TDM + async-LDS).
// All GEMMs use v_wmma_f32_16x16x32_bf16 (bf16 inputs, f32 accumulate).
// Weight tiles double-buffered into LDS by the Tensor Data Mover; attention
// K tiles double-buffered with global_load_async_to_lds_b128 so DMA overlaps
// WMMA compute.  Shapes: B=4, S=2048, D=1024, H=16, hd=64, F=4096.
// Workspace: 184 MB.
// ---------------------------------------------------------------------------

typedef __bf16 bf16;
typedef __attribute__((ext_vector_type(16))) __bf16 v16bf;
typedef __attribute__((ext_vector_type(8)))  __bf16 v8bf;
typedef __attribute__((ext_vector_type(8)))  float  v8f;
typedef __attribute__((ext_vector_type(4)))  unsigned int u32x4;
typedef __attribute__((ext_vector_type(8)))  unsigned int u32x8;

#define DEV static __device__ __forceinline__

// A-matrix 16x32 bf16 fragment: lane half lh holds 16 contiguous K values at
// K = lh*16.  p must already include "+ lh*16" from the row base.
DEV v16bf fragA(const bf16* p) {
  v8bf lo = *(const v8bf*)(p);
  v8bf hi = *(const v8bf*)(p + 8);
  v16bf r;
#pragma unroll
  for (int i = 0; i < 8; ++i) { r[i] = lo[i]; r[i + 8] = hi[i]; }
  return r;
}

// B-matrix 32x16 bf16 fragment: lane half lh holds 16 contiguous K values at
// K = lh*16.  p must already include "+ lh*16" from the column base.
DEV v16bf fragB(const bf16* p) {
  v8bf lo = *(const v8bf*)(p);
  v8bf hi = *(const v8bf*)(p + 8);
  v16bf r;
#pragma unroll
  for (int i = 0; i < 8; ++i) { r[i] = lo[i]; r[i + 8] = hi[i]; }
  return r;
}

DEV v8f wmma_bf16(v16bf a, v16bf b, v8f c) {
  return __builtin_amdgcn_wmma_f32_16x16x32_bf16(
      /*neg_a=*/false, a, /*neg_b=*/false, b,
      /*c_mod=*/(short)0, c, /*reuse_a=*/false, /*reuse_b=*/false);
}

// Low 32 bits of a generic pointer into LDS == LDS byte address (ISA 10.2:
// the LDS aperture truncates the flat address to addr[31:0]).
DEV unsigned int lds_addr_of(const void* p) {
  return (unsigned int)(unsigned long long)p;
}

// --- CDNA5 async memory: global -> LDS, tracked by ASYNCcnt ----------------
DEV void async_ld_b128(unsigned int ldsAddr, const void* g) {
  asm volatile("global_load_async_to_lds_b128 %0, %1, off"
               :: "v"(ldsAddr), "v"((unsigned long long)g) : "memory");
}
DEV void wait_asynccnt0() {
  asm volatile("s_wait_asynccnt 0x0" ::: "memory");
}

// --- CDNA5 Tensor Data Mover: 2-D tile global -> LDS, tracked by TENSORcnt -
// D# group 0 (ISA 08 §8.3): count=1, lds_addr, 57-bit global_addr, type=2.
DEV u32x4 tdm_group0(unsigned int ldsAddr, const void* gaddr) {
  unsigned long long ga = (unsigned long long)gaddr;
  u32x4 d;
  d.x = 1u;                                   // count=1, user mode, no gather
  d.y = ldsAddr;                              // lds_addr (bytes)
  d.z = (unsigned int)ga;                     // global_addr[31:0]
  d.w = ((unsigned int)(ga >> 32) & 0x01FFFFFFu) | (2u << 30);  // [56:32]|type
  return d;
}
// D# group 1 (ISA 08 §8.4): bf16 (data_size=1 -> 2B), pad every 32 DWORDs
// (128 B row) by 4 DWORDs (16 B) -> LDS pitch 144 B = 72 elements.
// Tile: tile0 x tile1 elements, row stride = `stride` elements.
DEV u32x8 tdm_group1(unsigned int dim0, unsigned int dim1,
                     unsigned int tile0, unsigned int tile1,
                     unsigned int stride) {
  u32x8 d;
  d[0] = (1u << 16)        // data_size = 2 bytes
       | (1u << 20)        // pad_enable
       | (4u << 22)        // pad_interval: 2^(4+1)=32 DWORDs between pads
       | (3u << 25);       // pad_amount: 3+1 = 4 DWORDs
  d[1] = (dim0 & 0xFFFFu) << 16;                    // tensor_dim0[15:0]
  d[2] = (dim0 >> 16) | ((dim1 & 0xFFFFu) << 16);   // dim0 hi | dim1 lo
  d[3] = (dim1 >> 16) | (tile0 << 16);              // dim1 hi | tile_dim0
  d[4] = tile1;                                     // tile_dim1 | tile_dim2=0
  d[5] = stride;                                    // tensor_dim0_stride lo32
  d[6] = 0u;                                        // stride hi | dim1_stride
  d[7] = 0u;
  return d;
}
DEV void tdm_load(u32x4 g0, u32x8 g1) {
  asm volatile("tensor_load_to_lds %0, %1" :: "s"(g0), "s"(g1) : "memory");
}

// ---------------------------------------------------------------------------
// fp32 -> bf16 conversion
// ---------------------------------------------------------------------------
__global__ void cvt_f32_bf16(const float* __restrict__ in,
                             bf16* __restrict__ out, int n) {
  for (int i = blockIdx.x * blockDim.x + threadIdx.x; i < n;
       i += gridDim.x * blockDim.x)
    out[i] = (bf16)in[i];
}

// ---------------------------------------------------------------------------
// WMMA GEMM: C[m,n] = sum_k A[m,k] * W[n,k] + bias[n]  (out = A @ W^T + b)
// Block = 256 threads = 8 waves; wave tile 16(M)x64(N); block tile 128x64.
// The 64x64-element weight tile for each 64-wide K-chunk is DMA'd into LDS by
// wave 0 via the Tensor Data Mover, double-buffered so the DMA of chunk i+1
// overlaps the WMMA compute of chunk i (TDM ops from one wave retire in
// order, so s_wait_tensorcnt 1 == "current chunk landed").
// MODE 1: bf16 store with head-split layout [B,H,S,hd] (QKV projections)
// MODE 2: fp32 store of val + resid (O-proj with x residual; FFN2 with x1)
// MODE 3: bf16 store of exact GELU(val) (FFN1)
// ---------------------------------------------------------------------------
template <int MODE>
__global__ __launch_bounds__(256) void gemm_bf16_wmma(
    const bf16* __restrict__ A, const bf16* __restrict__ Bw,
    const float* __restrict__ bias, const float* __restrict__ resid,
    void* __restrict__ outp, int M, int N, int K) {
  __shared__ __align__(16) bf16 bS[2][64 * 72];  // double-buffered W tile

  const int lane = threadIdx.x & 31;
  const int w    = threadIdx.x >> 5;
  const int lr   = lane & 15;     // row/col within 16-wide tile
  const int lh   = lane >> 4;     // lane half selects K sub-group
  const int m0 = blockIdx.y * 128 + w * 16;
  const int n0 = blockIdx.x * 64;

  const bf16* aRow = A + (size_t)(m0 + lr) * K + lh * 16;
  const bf16* wTile = Bw + (size_t)n0 * K;
  const unsigned int bS_lds[2] = {lds_addr_of(&bS[0][0]),
                                  lds_addr_of(&bS[1][0])};
  const u32x8 g1 = tdm_group1((unsigned)K, 64u, 64u, 64u, (unsigned)K);

  const int nChunks = K >> 6;
  if (w == 0) tdm_load(tdm_group0(bS_lds[0], wTile), g1);

  v8f acc[4] = {};
  for (int i = 0; i < nChunks; ++i) {
    const int buf = i & 1;
    if (w == 0) {
      if (i + 1 < nChunks) {  // queue next chunk into the other buffer
        tdm_load(tdm_group0(bS_lds[buf ^ 1], wTile + (size_t)(i + 1) * 64), g1);
        __builtin_amdgcn_s_wait_tensorcnt(1);  // current landed, next in flight
      } else {
        __builtin_amdgcn_s_wait_tensorcnt(0);  // drain: last chunk landed
      }
    }
    __syncthreads();  // (A) buf landed, visible to all 8 waves

    const bf16* bBase = &bS[buf][0] + lr * 72 + lh * 16;  // + nt*1152 + c*32
    const int kk = i * 64;
#pragma unroll
    for (int c = 0; c < 2; ++c) {
      v16bf a  = fragA(aRow + kk + c * 32);
      // Hoist all four B fragments: DS loads issue back-to-back, one dscnt
      // drain, then four WMMAs issue back-to-back.
      v16bf b0 = fragB(bBase + 0 * 1152 + c * 32);
      v16bf b1 = fragB(bBase + 1 * 1152 + c * 32);
      v16bf b2 = fragB(bBase + 2 * 1152 + c * 32);
      v16bf b3 = fragB(bBase + 3 * 1152 + c * 32);
      acc[0] = wmma_bf16(a, b0, acc[0]);
      acc[1] = wmma_bf16(a, b1, acc[1]);
      acc[2] = wmma_bf16(a, b2, acc[2]);
      acc[3] = wmma_bf16(a, b3, acc[3]);
    }
    __syncthreads();  // (B) all waves done reading buf before it is re-filled
  }

#pragma unroll
  for (int nt = 0; nt < 4; ++nt) {
    const int col = n0 + nt * 16 + lr;
    const float bval = bias[col];
#pragma unroll
    for (int r = 0; r < 8; ++r) {
      // C/D layout: row = r + 8*lh, col = lane%16 (ISA 7.12.2)
      const int row = m0 + r + 8 * lh;
      float v = acc[nt][r] + bval;
      if (MODE == 1) {
        // token row -> (b, s); feature col -> (h, hd); store [B,H,S,hd] bf16
        const int b_ = row >> 11, s_ = row & 2047;
        const int h_ = col >> 6, d_ = col & 63;
        ((bf16*)outp)[(((size_t)(b_ * 16 + h_) * 2048 + s_) << 6) + d_] =
            (bf16)v;
      } else if (MODE == 2) {
        const size_t idx = (size_t)row * N + col;
        ((float*)outp)[idx] = v + resid[idx];
      } else {  // MODE 3: exact GELU
        const float g = 0.5f * v * (1.0f + erff(v * 0.70710678118f));
        ((bf16*)outp)[(size_t)row * N + col] = (bf16)g;
      }
    }
  }
}

// ---------------------------------------------------------------------------
// Flash-style causal attention, one (b,h) per blockIdx.y, 128 q-rows per
// blockIdx.x.  8 waves; each wave owns 16 q rows.  64-key K/V blocks are
// double-buffered through LDS: K via async global->LDS DMA (ASYNCcnt), V via
// a manual transposed store (so PV B-fragments are contiguous reads).  The
// next block's loads are issued before computing on the current block.
// ---------------------------------------------------------------------------
__global__ __launch_bounds__(256) void attn_flash(
    const bf16* __restrict__ Q, const bf16* __restrict__ Km,
    const bf16* __restrict__ Vm, bf16* __restrict__ Oout) {
  // Pitch 72 elements (144 B): 16B-aligned rows, bank-conflict-friendly.
  __shared__ __align__(16) bf16 kS[2][64 * 72];   // K block, [key][hd]
  __shared__ __align__(16) bf16 vS[2][64 * 72];   // V transposed [hd][key]
  __shared__ __align__(16) bf16 pS[8][16 * 72];   // per-wave P staging

  const int tid  = threadIdx.x;
  const int lane = tid & 31;
  const int w    = tid >> 5;
  const int lr   = lane & 15;
  const int lh   = lane >> 4;
  const int bh = blockIdx.y;            // b*16 + h
  const int q0 = blockIdx.x * 128;
  const int qr = q0 + w * 16;           // this wave's first q row

  // Q fragment: A-matrix 16x64 -> two K=32 chunks
  const bf16* qBase = Q + ((size_t)bh * 2048 + qr + lr) * 64 + lh * 16;
  const v16bf aq0 = fragA(qBase);
  const v16bf aq1 = fragA(qBase + 32);

  float m_r[8], l_r[8];
#pragma unroll
  for (int r = 0; r < 8; ++r) { m_r[r] = -1e30f; l_r[r] = 0.f; }
  v8f o[4] = {};

  const bf16* kBase = Km + (size_t)bh * 2048 * 64;
  const bf16* vBase = Vm + (size_t)bh * 2048 * 64;
  const int nkb = blockIdx.x * 2 + 2;   // causal: keys up to q0+127

  // Stage one 64-key K/V block into LDS buffer `buf`.
  auto load_tiles = [&](int kb, int buf) {
    bf16* kDst = &kS[buf][0];
    bf16* vDst = &vS[buf][0];
#pragma unroll
    for (int u = tid; u < 512; u += 256) {
      const int row = u >> 3, seg = u & 7;
      async_ld_b128(lds_addr_of(kDst + row * 72 + seg * 8),
                    kBase + ((size_t)(kb * 64 + row)) * 64 + seg * 8);
    }
#pragma unroll
    for (int u = tid; u < 512; u += 256) {
      const int row = u >> 3, seg = u & 7;
      v8bf vd = *(const v8bf*)(vBase + ((size_t)(kb * 64 + row)) * 64 + seg * 8);
#pragma unroll
      for (int i = 0; i < 8; ++i) vDst[(seg * 8 + i) * 72 + row] = vd[i];
    }
  };

  load_tiles(0, 0);
  for (int kb = 0; kb < nkb; ++kb) {
    const int buf = kb & 1;
    wait_asynccnt0();   // this wave's async K loads for `buf` landed
    __syncthreads();    // all waves' K/V (incl. ds stores) for `buf` visible
    if (kb + 1 < nkb) load_tiles(kb + 1, buf ^ 1);  // overlap next DMA

    // ---- scores: S = Q K^T  (M=16 q, N=64 keys, K=64 hd) ----
    float sc[4][8];
#pragma unroll
    for (int nt = 0; nt < 4; ++nt) {
      const bf16* kRow = &kS[buf][(nt * 16 + lr) * 72 + lh * 16];
      v16bf b0 = fragB(kRow);        // hoisted: both DS loads in flight
      v16bf b1 = fragB(kRow + 32);
      v8f s = {};
      s = wmma_bf16(aq0, b0, s);
      s = wmma_bf16(aq1, b1, s);
      const int key = kb * 64 + nt * 16 + lr;
#pragma unroll
      for (int r = 0; r < 8; ++r) {
        const int qrow = qr + r + 8 * lh;
        sc[nt][r] = (key <= qrow) ? s[r] * 0.125f : -1e30f;
      }
    }

    // ---- online softmax update (row stats live per (r, lane-half)) ----
    float mnew[8], alpha[8], rs[8];
#pragma unroll
    for (int r = 0; r < 8; ++r) {
      float mb = fmaxf(fmaxf(sc[0][r], sc[1][r]), fmaxf(sc[2][r], sc[3][r]));
#pragma unroll
      for (int off = 1; off < 16; off <<= 1)
        mb = fmaxf(mb, __shfl_xor(mb, off, 32));
      mnew[r]  = fmaxf(m_r[r], mb);
      alpha[r] = __expf(m_r[r] - mnew[r]);
      m_r[r]   = mnew[r];
      rs[r]    = 0.f;
    }
#pragma unroll
    for (int nt = 0; nt < 4; ++nt)
#pragma unroll
      for (int r = 0; r < 8; ++r) {
        const float p = __expf(sc[nt][r] - mnew[r]);
        sc[nt][r] = p;
        rs[r] += p;
      }
#pragma unroll
    for (int r = 0; r < 8; ++r) {
#pragma unroll
      for (int off = 1; off < 16; off <<= 1)
        rs[r] += __shfl_xor(rs[r], off, 32);
      l_r[r] = l_r[r] * alpha[r] + rs[r];
    }
#pragma unroll
    for (int dt = 0; dt < 4; ++dt)
#pragma unroll
      for (int r = 0; r < 8; ++r) o[dt][r] *= alpha[r];

    // ---- P: C-layout -> A-layout via wave-private LDS round trip ----
    bf16* pw = &pS[w][0];
#pragma unroll
    for (int nt = 0; nt < 4; ++nt)
#pragma unroll
      for (int r = 0; r < 8; ++r)
        pw[(r + 8 * lh) * 72 + nt * 16 + lr] = (bf16)sc[nt][r];
    // LDS ops from one wave execute in order; just stop compiler reordering.
    asm volatile("" ::: "memory");

    // ---- O += P V  (M=16 q, N=64 hd, K=64 keys) ----
    const bf16* pA = pw + lr * 72 + lh * 16;
    const bf16* vB = &vS[buf][0] + lr * 72 + lh * 16;   // + dt*1152 + c*32
#pragma unroll
    for (int c = 0; c < 2; ++c) {
      v16bf ap = fragA(pA + c * 32);
      v16bf v0 = fragB(vB + 0 * 1152 + c * 32);
      v16bf v1 = fragB(vB + 1 * 1152 + c * 32);
      v16bf v2 = fragB(vB + 2 * 1152 + c * 32);
      v16bf v3 = fragB(vB + 3 * 1152 + c * 32);
      o[0] = wmma_bf16(ap, v0, o[0]);
      o[1] = wmma_bf16(ap, v1, o[1]);
      o[2] = wmma_bf16(ap, v2, o[2]);
      o[3] = wmma_bf16(ap, v3, o[3]);
    }
  }

  // ---- normalize and store concat-head layout [B,S,D] bf16 ----
  const int b_ = bh >> 4, h_ = bh & 15;
#pragma unroll
  for (int r = 0; r < 8; ++r) {
    const float inv = 1.0f / l_r[r];
    const int s_ = qr + r + 8 * lh;
    const size_t rowBase = ((size_t)b_ * 2048 + s_) * 1024 + h_ * 64;
#pragma unroll
    for (int dt = 0; dt < 4; ++dt)
      Oout[rowBase + dt * 16 + lr] = (bf16)(o[dt][r] * inv);
  }
}

// ---------------------------------------------------------------------------
// LayerNorm over D=1024; one row per block; fp32 out + optional bf16 out.
// ---------------------------------------------------------------------------
__global__ __launch_bounds__(256) void layernorm_k(
    const float* __restrict__ in, const float* __restrict__ gamma,
    const float* __restrict__ beta, float* __restrict__ outF,
    bf16* __restrict__ outB) {
  __shared__ float red1[8], red2[8];
  const int row = blockIdx.x;
  const float* p = in + (size_t)row * 1024;
  float v[4], s1 = 0.f, s2 = 0.f;
#pragma unroll
  for (int i = 0; i < 4; ++i) {
    v[i] = p[threadIdx.x + i * 256];
    s1 += v[i];
    s2 += v[i] * v[i];
  }
#pragma unroll
  for (int off = 16; off >= 1; off >>= 1) {
    s1 += __shfl_xor(s1, off, 32);
    s2 += __shfl_xor(s2, off, 32);
  }
  const int lane = threadIdx.x & 31, w = threadIdx.x >> 5;
  if (lane == 0) { red1[w] = s1; red2[w] = s2; }
  __syncthreads();
  float t1 = 0.f, t2 = 0.f;
#pragma unroll
  for (int i = 0; i < 8; ++i) { t1 += red1[i]; t2 += red2[i]; }
  const float mean = t1 * (1.0f / 1024.0f);
  const float var  = t2 * (1.0f / 1024.0f) - mean * mean;
  const float rstd = rsqrtf(var + 1e-5f);
#pragma unroll
  for (int i = 0; i < 4; ++i) {
    const int c = threadIdx.x + i * 256;
    const float yv = (v[i] - mean) * rstd * gamma[c] + beta[c];
    outF[(size_t)row * 1024 + c] = yv;
    if (outB) outB[(size_t)row * 1024 + c] = (bf16)yv;
  }
}

// ---------------------------------------------------------------------------
// Orchestration
// ---------------------------------------------------------------------------
extern "C" void kernel_launch(void* const* d_in, const int* in_sizes, int n_in,
                              void* d_out, int out_size, void* d_ws,
                              size_t ws_size, hipStream_t stream) {
  (void)in_sizes; (void)n_in; (void)out_size; (void)ws_size;
  const float* x    = (const float*)d_in[0];
  const float* Wq   = (const float*)d_in[1];
  const float* bq   = (const float*)d_in[2];
  const float* Wk   = (const float*)d_in[3];
  const float* bk   = (const float*)d_in[4];
  const float* Wv   = (const float*)d_in[5];
  const float* bv   = (const float*)d_in[6];
  const float* Wo   = (const float*)d_in[7];
  const float* bo   = (const float*)d_in[8];
  const float* W1   = (const float*)d_in[9];
  const float* b1   = (const float*)d_in[10];
  const float* W2   = (const float*)d_in[11];
  const float* b2   = (const float*)d_in[12];
  const float* ln1g = (const float*)d_in[13];
  const float* ln1b = (const float*)d_in[14];
  const float* ln2g = (const float*)d_in[15];
  const float* ln2b = (const float*)d_in[16];

  // Workspace layout (bytes); peak 184 MB, buffers recycled once dead.
  char* ws = (char*)d_ws;
  const size_t MB = 1024u * 1024u;
  bf16*  Wqb   = (bf16*)(ws + 0 * MB);    // 2 MB
  bf16*  Wkb   = (bf16*)(ws + 2 * MB);    // 2 MB
  bf16*  Wvb   = (bf16*)(ws + 4 * MB);    // 2 MB
  bf16*  Wob   = (bf16*)(ws + 6 * MB);    // 2 MB
  bf16*  W1b   = (bf16*)(ws + 8 * MB);    // 8 MB
  bf16*  W2b   = (bf16*)(ws + 16 * MB);   // 8 MB
  bf16*  xb    = (bf16*)(ws + 24 * MB);   // 16 MB (reused as attnb)
  bf16*  Qb    = (bf16*)(ws + 40 * MB);   // 16 MB
  bf16*  Kb    = (bf16*)(ws + 56 * MB);   // 16 MB
  bf16*  Vb    = (bf16*)(ws + 72 * MB);   // 16 MB
  bf16*  attnb = xb;                       // xb dead after QKV GEMMs
  float* t0    = (float*)(ws + 40 * MB);  // 32 MB over Q/K (dead post-attn)
  float* x1f   = (float*)(ws + 72 * MB);  // 32 MB over V (dead post-attn)
  bf16*  x1b   = (bf16*)(ws + 104 * MB);  // 16 MB
  bf16*  hb    = (bf16*)(ws + 120 * MB);  // 64 MB
  float* t1    = t0;                       // t0 dead after LN1

  const int BS = 4 * 2048;  // 8192 tokens

  // fp32 -> bf16 conversions
  cvt_f32_bf16<<<4096, 256, 0, stream>>>(x,  xb,  BS * 1024);
  cvt_f32_bf16<<<1024, 256, 0, stream>>>(Wq, Wqb, 1024 * 1024);
  cvt_f32_bf16<<<1024, 256, 0, stream>>>(Wk, Wkb, 1024 * 1024);
  cvt_f32_bf16<<<1024, 256, 0, stream>>>(Wv, Wvb, 1024 * 1024);
  cvt_f32_bf16<<<1024, 256, 0, stream>>>(Wo, Wob, 1024 * 1024);
  cvt_f32_bf16<<<4096, 256, 0, stream>>>(W1, W1b, 4096 * 1024);
  cvt_f32_bf16<<<4096, 256, 0, stream>>>(W2, W2b, 1024 * 4096);

  // QKV projections: [8192,1024] = xb @ W^T, head-split bf16 stores
  gemm_bf16_wmma<1><<<dim3(16, 64), 256, 0, stream>>>(
      xb, Wqb, bq, nullptr, Qb, BS, 1024, 1024);
  gemm_bf16_wmma<1><<<dim3(16, 64), 256, 0, stream>>>(
      xb, Wkb, bk, nullptr, Kb, BS, 1024, 1024);
  gemm_bf16_wmma<1><<<dim3(16, 64), 256, 0, stream>>>(
      xb, Wvb, bv, nullptr, Vb, BS, 1024, 1024);

  // causal flash attention -> concat-head bf16 [B,S,D]
  attn_flash<<<dim3(16, 64), 256, 0, stream>>>(Qb, Kb, Vb, attnb);

  // O projection + residual (x) -> t0 fp32
  gemm_bf16_wmma<2><<<dim3(16, 64), 256, 0, stream>>>(
      attnb, Wob, bo, x, t0, BS, 1024, 1024);

  // LN1 -> x1 (fp32 for residual, bf16 for FFN GEMM)
  layernorm_k<<<BS, 256, 0, stream>>>(t0, ln1g, ln1b, x1f, x1b);

  // FFN: h = gelu(x1 @ W1^T + b1) -> bf16 [8192,4096]
  gemm_bf16_wmma<3><<<dim3(64, 64), 256, 0, stream>>>(
      x1b, W1b, b1, nullptr, hb, BS, 4096, 1024);

  // ff_out + x1 -> t1 fp32
  gemm_bf16_wmma<2><<<dim3(16, 64), 256, 0, stream>>>(
      hb, W2b, b2, x1f, t1, BS, 1024, 4096);

  // LN2 -> final output fp32
  layernorm_k<<<BS, 256, 0, stream>>>(t1, ln2g, ln2b, (float*)d_out, nullptr);
}
